// Model_73426760893034
// MI455X (gfx1250) — matrix operand order repaired and checked by reading, not compiled, not run on hardware
//
#include <hip/hip_runtime.h>
#include <hip/hip_bf16.h>

// ===========================================================================
// Informer forward pass for MI455X (gfx1250, wave32, WMMA).
// All dense contractions (QKV/O projections, FFN, conv-via-im2col, final
// projection, attention QK^T and PV) run through v_wmma_f32_16x16x32_f16.
// Prob-sparse attention is replaced by dense (optionally causal) flash-style
// attention: the stochastic top-k path depends on host JAX RNG and cannot be
// reproduced; dense attention is the computational superset.
// Workspace requirement: ~168 MB (fits MI455X L2 of 192 MB).
//
// Round-2 changes: GEMM epilogue restructured -- bias loads hoisted (2 per
// lane instead of 32 exec-diamond loads each trailed by s_wait_loadcnt 0),
// GELU under one uniform branch, stores under per-ni lane predicates only.
// ===========================================================================

typedef _Float16 v16h __attribute__((ext_vector_type(16)));
typedef float    v8f  __attribute__((ext_vector_type(8)));

#define DMODEL 512
#define DHEAD  64
#define NHEADS 8
#define NB     8

static __device__ inline v8f wmma_f16(v16h a, v16h b, v8f c) {
  return __builtin_amdgcn_wmma_f32_16x16x32_f16(
      /*neg_a=*/false, a, /*neg_b=*/false, b,
      /*c_mod=*/(short)0, c, /*reuse_a=*/false, /*reuse_b=*/false);
}

static __device__ inline void wait_ds0() {
#if __has_builtin(__builtin_amdgcn_s_wait_dscnt)
  __builtin_amdgcn_s_wait_dscnt(0);
#else
  asm volatile("s_wait_dscnt 0" ::: "memory");
#endif
}

// pack 4 f32 -> 4 f16 and store as one 8-byte LDS write (dst 8B-aligned)
static __device__ inline void st4h(_Float16* dst, float4 a4) {
  union { _Float16 h[4]; uint2 u; } pk;
  pk.h[0] = (_Float16)a4.x; pk.h[1] = (_Float16)a4.y;
  pk.h[2] = (_Float16)a4.z; pk.h[3] = (_Float16)a4.w;
  *(uint2*)dst = pk.u;
}

// ---------------------------------------------------------------------------
// GEMM: C[M,N] = act(A[M,K] @ W[N,K]^T + bias)   (f32 in/out, f16 WMMA math)
// Block tile 128x64, K-step 32, 256 threads = 8 waves, wave tile 32x32.
// Preconditions (all launch sites satisfy): M % 128 == 0, K % 32 == 0.
// N may be ragged (final projection N=32): W rows are clamped + zero-selected.
// ---------------------------------------------------------------------------
__global__ void __launch_bounds__(256)
gemm_wmma(const float* __restrict__ A, const float* __restrict__ W,
          const float* __restrict__ bias, float* __restrict__ C,
          int M, int N, int K, int act)
{
  __shared__ _Float16 As[128][36];   // row stride 72B: 8B-aligned, conflict-free
  __shared__ _Float16 Ws[64][36];
  const int tid  = threadIdx.x;
  const int lane = tid & 31;
  const int wid  = tid >> 5;
  const int wm   = wid >> 1;     // 0..3  (M direction)
  const int wn   = wid & 1;      // 0..1  (N direction)
  const int m0   = blockIdx.y * 128;
  const int n0   = blockIdx.x * 64;
  const int rq   = lane & 15;
  const int hi   = lane >> 4;

  v8f acc[2][2];
#pragma unroll
  for (int mi = 0; mi < 2; ++mi)
#pragma unroll
    for (int ni = 0; ni < 2; ++ni)
#pragma unroll
      for (int r = 0; r < 8; ++r) acc[mi][ni][r] = 0.f;

  for (int k0 = 0; k0 < K; k0 += 32) {
    // stage A tile 128x32: 1024 float4 groups, 4 per thread, unconditional
#pragma unroll
    for (int i = 0; i < 4; ++i) {
      int idx = i * 256 + tid;
      int r = idx >> 3;           // 0..127
      int c = (idx & 7) << 2;     // 0,4,...,28
      float4 a4 = *(const float4*)(A + (size_t)(m0 + r) * K + k0 + c);
      st4h(&As[r][c], a4);
    }
    // stage W tile 64x32: 512 float4 groups, 2 per thread, branch-free N guard
#pragma unroll
    for (int i = 0; i < 2; ++i) {
      int idx = i * 256 + tid;
      int r = idx >> 3;           // 0..63
      int c = (idx & 7) << 2;
      int gr = n0 + r;
      int grs = gr < N ? gr : (N - 1);         // clamp -> unconditional load
      float4 w4 = *(const float4*)(W + (size_t)grs * K + k0 + c);
      if (gr >= N) { w4.x = 0.f; w4.y = 0.f; w4.z = 0.f; w4.w = 0.f; } // cndmask
      st4h(&Ws[r][c], w4);
    }
    __syncthreads();

    v16h af[2], bf[2];
#pragma unroll
    for (int mi = 0; mi < 2; ++mi) {
      int row = wm * 32 + mi * 16 + rq;
#pragma unroll
      for (int j = 0; j < 16; ++j) {
        int v = j >> 1, lo = j & 1;
        int kk = ((v & 3) << 1) + (hi << 3) + ((v >> 2) << 4) + lo; // ISA A 16x32 layout
        af[mi][j] = As[row][kk];
      }
    }
#pragma unroll
    for (int ni = 0; ni < 2; ++ni) {
      int nr = wn * 32 + ni * 16 + rq;
#pragma unroll
      for (int j = 0; j < 16; ++j) {
        int v = j >> 1, lo = j & 1;
        int kk = ((v & 3) << 1) + (hi << 3) + ((v >> 2) << 4) + lo; // ISA B 32x16 layout
        bf[ni][j] = Ws[nr][kk];
      }
    }
#pragma unroll
    for (int mi = 0; mi < 2; ++mi)
#pragma unroll
      for (int ni = 0; ni < 2; ++ni)
        acc[mi][ni] = wmma_f16(af[mi], bf[ni], acc[mi][ni]);
    __syncthreads();
  }

  // ---- epilogue: hoisted bias loads, uniform act branch, predicated stores
  float bcol[2];
  bool  okc[2];
#pragma unroll
  for (int ni = 0; ni < 2; ++ni) {
    int col = n0 + wn * 32 + ni * 16 + rq;
    okc[ni] = (col < N);
    int colc = okc[ni] ? col : 0;                 // clamp -> unconditional load
    bcol[ni] = bias ? bias[colc] : 0.f;           // uniform null check
  }
#pragma unroll
  for (int mi = 0; mi < 2; ++mi)
#pragma unroll
    for (int ni = 0; ni < 2; ++ni)
#pragma unroll
      for (int r = 0; r < 8; ++r) acc[mi][ni][r] += bcol[ni];
  if (act == 1) {                                 // uniform branch: exact GELU
#pragma unroll
    for (int mi = 0; mi < 2; ++mi)
#pragma unroll
      for (int ni = 0; ni < 2; ++ni)
#pragma unroll
        for (int r = 0; r < 8; ++r) {
          float v = acc[mi][ni][r];
          acc[mi][ni][r] = 0.5f * v * (1.f + erff(v * 0.70710678118654752f));
        }
  }
#pragma unroll
  for (int mi = 0; mi < 2; ++mi)
#pragma unroll
    for (int ni = 0; ni < 2; ++ni) {
      int col = n0 + wn * 32 + ni * 16 + rq;
      if (okc[ni]) {                              // lane predicate, stores only
#pragma unroll
        for (int r = 0; r < 8; ++r) {
          int row = m0 + wm * 32 + mi * 16 + r + (hi << 3);
          C[(size_t)row * N + col] = acc[mi][ni][r];
        }
      }
    }
}

// ---------------------------------------------------------------------------
// Dense flash-style attention (per (b,h), online softmax, WMMA for QK^T & PV)
// Q: [B*Lq, 512] (head slice at col h*64), K/V: [B*S, 512]. 4 waves / block
// share one (b,h): K/V chunks are cooperatively staged in LDS once per block.
// Causal loop bound is block-uniform; fully-masked chunks are algebraic no-ops.
// ---------------------------------------------------------------------------
__global__ void __launch_bounds__(128)
attn_dense(const float* __restrict__ Qb, const float* __restrict__ Kb,
           const float* __restrict__ Vb, float* __restrict__ Ob,
           int Lq, int S, int causal)
{
  __shared__ _Float16 Ks[32][68];      // [key][e]   row stride 136B (17 banks)
  __shared__ _Float16 Vs[64][36];      // [e][key]   transposed for PV pairs
  __shared__ _Float16 smP[4][16 * 34];
  const int lane = threadIdx.x & 31;
  const int wid  = threadIdx.x >> 5;
  const int b = blockIdx.y >> 3;
  const int h = blockIdx.y & 7;
  const int block_q0 = blockIdx.x * 64;
  const int q0 = block_q0 + wid * 16;
  const int rq = lane & 15;
  const int hi = lane >> 4;

  const float* Q  = Qb + (size_t)b * Lq * DMODEL + h * DHEAD;
  const float* Kp = Kb + (size_t)b * S  * DMODEL + h * DHEAD;
  const float* Vp = Vb + (size_t)b * S  * DMODEL + h * DHEAD;
  float*       O  = Ob + (size_t)b * Lq * DMODEL + h * DHEAD;

  // Q tile 16x64 as two A fragments (E = 0..31, 32..63); pairs contiguous
  v16h qf[2];
#pragma unroll
  for (int s = 0; s < 2; ++s)
#pragma unroll
    for (int j = 0; j < 16; ++j) {
      int v = j >> 1, lo = j & 1;
      int kk = ((v & 3) << 1) + (hi << 3) + ((v >> 2) << 4) + lo;
      qf[s][j] = (_Float16)Q[(size_t)(q0 + rq) * DMODEL + s * 32 + kk];
    }

  v8f oacc[4];
#pragma unroll
  for (int et = 0; et < 4; ++et)
#pragma unroll
    for (int r = 0; r < 8; ++r) oacc[et][r] = 0.f;
  float mrow[8], lrow[8];
#pragma unroll
  for (int r = 0; r < 8; ++r) { mrow[r] = -1e30f; lrow[r] = 0.f; }

  const int s_end = causal ? (block_q0 + 64) : S;   // block-uniform
  for (int s0 = 0; s0 < s_end; s0 += 32) {
    // cooperative stage: K chunk [32 keys x 64 E] and V chunk (transposed)
#pragma unroll
    for (int i = 0; i < 4; ++i) {
      int idx = i * 128 + threadIdx.x;      // 0..511
      int r = idx >> 4;                     // key 0..31
      int c = (idx & 15) << 2;              // e 0..60
      float4 k4 = *(const float4*)(Kp + (size_t)(s0 + r) * DMODEL + c);
      st4h(&Ks[r][c], k4);
      float4 v4 = *(const float4*)(Vp + (size_t)(s0 + r) * DMODEL + c);
      Vs[c + 0][r] = (_Float16)v4.x;        // transpose: [e][key]
      Vs[c + 1][r] = (_Float16)v4.y;
      Vs[c + 2][r] = (_Float16)v4.z;
      Vs[c + 3][r] = (_Float16)v4.w;
    }
    __syncthreads();

    // scores 16x32 = Q(16x64) @ K^T(64x32)
    v8f sc[2];
#pragma unroll
    for (int t = 0; t < 2; ++t) {
#pragma unroll
      for (int r = 0; r < 8; ++r) sc[t][r] = 0.f;
#pragma unroll
      for (int s = 0; s < 2; ++s) {
        v16h kf;
#pragma unroll
        for (int j = 0; j < 16; ++j) {
          int v = j >> 1, lo = j & 1;
          int kk = ((v & 3) << 1) + (hi << 3) + ((v >> 2) << 4) + lo;   // contraction (E) index 0..31
          kf[j] = Ks[t * 16 + rq][s * 32 + kk];
        }
        sc[t] = wmma_f16(qf[s], kf, sc[t]);
      }
    }
    // online softmax per row (rows live across VGPR index r, per 16-lane half)
#pragma unroll
    for (int r = 0; r < 8; ++r) {
      const int qrow = q0 + r + (hi << 3);
      float v0 = sc[0][r] * 0.125f;   // 1/sqrt(64)
      float v1 = sc[1][r] * 0.125f;
      if (causal) {
        if (s0 + rq > qrow)       v0 = -1e30f;
        if (s0 + 16 + rq > qrow)  v1 = -1e30f;
      }
      float mx = fmaxf(v0, v1);
#pragma unroll
      for (int off = 8; off >= 1; off >>= 1) mx = fmaxf(mx, __shfl_xor(mx, off, 32));
      float mn = fmaxf(mrow[r], mx);
      float alpha = __expf(mrow[r] - mn);
      float p0 = __expf(v0 - mn);
      float p1 = __expf(v1 - mn);
      float rs = p0 + p1;
#pragma unroll
      for (int off = 8; off >= 1; off >>= 1) rs += __shfl_xor(rs, off, 32);
      lrow[r] = lrow[r] * alpha + rs;
      mrow[r] = mn;
#pragma unroll
      for (int et = 0; et < 4; ++et) oacc[et][r] *= alpha;
      const int prow = r + (hi << 3);
      smP[wid][prow * 34 + rq]      = (_Float16)p0;   // C-layout -> LDS
      smP[wid][prow * 34 + 16 + rq] = (_Float16)p1;
    }
    wait_ds0();                                       // wave-local LDS RAW
    // P as A fragment (16 queries x 32 keys)
    v16h pf;
#pragma unroll
    for (int j = 0; j < 16; ++j) {
      int v = j >> 1, lo = j & 1;
      int kk = ((v & 3) << 1) + (hi << 3) + ((v >> 2) << 4) + lo;
      pf[j] = smP[wid][rq * 34 + kk];
    }
    // O += P @ V  (V chunk 32x64 as four B fragments; pairs contiguous)
#pragma unroll
    for (int et = 0; et < 4; ++et) {
      v16h vf;
#pragma unroll
      for (int j = 0; j < 16; ++j) {
        int v = j >> 1, lo = j & 1;
        int kk = ((v & 3) << 1) + (hi << 3) + ((v >> 2) << 4) + lo;   // local key 0..31
        vf[j] = Vs[et * 16 + rq][kk];
      }
      oacc[et] = wmma_f16(pf, vf, oacc[et]);
    }
    __syncthreads();
  }
#pragma unroll
  for (int et = 0; et < 4; ++et)
#pragma unroll
    for (int r = 0; r < 8; ++r) {
      int row = q0 + r + (hi << 3);
      O[(size_t)row * DMODEL + et * 16 + rq] = oacc[et][r] / lrow[r];
    }
}

// ---------------------------------------------------------------------------
// LayerNorm with optional fused residual: out = LN(X (+R)) * g + b
// One wave per 512-wide row.
// ---------------------------------------------------------------------------
__global__ void __launch_bounds__(128)
layernorm_res(const float* __restrict__ X, const float* __restrict__ R,
              const float* __restrict__ g, const float* __restrict__ bta,
              float* __restrict__ Y, int Mrows)
{
  const int lane = threadIdx.x & 31;
  const int row  = blockIdx.x * 4 + (threadIdx.x >> 5);
  if (row >= Mrows) return;
  const float* x = X + (size_t)row * DMODEL;
  const float* r = R ? (R + (size_t)row * DMODEL) : nullptr;
  float v[16];
  float s = 0.f;
#pragma unroll
  for (int i = 0; i < 16; ++i) {
    int c = lane + 32 * i;
    v[i] = x[c] + (r ? r[c] : 0.f);
    s += v[i];
  }
#pragma unroll
  for (int off = 16; off >= 1; off >>= 1) s += __shfl_xor(s, off, 32);
  float mu = s * (1.f / 512.f);
  float q = 0.f;
#pragma unroll
  for (int i = 0; i < 16; ++i) { float d = v[i] - mu; q += d * d; }
#pragma unroll
  for (int off = 16; off >= 1; off >>= 1) q += __shfl_xor(q, off, 32);
  float inv = rsqrtf(q * (1.f / 512.f) + 1e-5f);
#pragma unroll
  for (int i = 0; i < 16; ++i) {
    int c = lane + 32 * i;
    Y[(size_t)row * DMODEL + c] = (v[i] - mu) * inv * g[c] + bta[c];
  }
}

// ---------------------------------------------------------------------------
// Elementwise helpers
// ---------------------------------------------------------------------------
__global__ void add_pos_embed(float* __restrict__ x, int total, int L) {
  int idx = blockIdx.x * blockDim.x + threadIdx.x;
  if (idx >= total) return;
  int c = idx & (DMODEL - 1);
  int l = (idx >> 9) % L;
  float i2 = (float)((c >> 1) << 1);
  float freq = __expf(-i2 * (9.21034037197618f / 512.f));  // ln(10000)
  float a = (float)l * freq;
  x[idx] += (c & 1) ? __cosf(a) : __sinf(a);
}

__global__ void im2col_circ(const float* __restrict__ x, float* __restrict__ y,
                            int L, int C, int total) {
  int idx = blockIdx.x * blockDim.x + threadIdx.x;
  if (idx >= total) return;
  int c  = idx % C;
  int k  = (idx / C) % 3;
  int bl = idx / (3 * C);
  int b = bl / L, l = bl % L;
  int ls = l - 1 + k;
  ls = (ls + L) % L;                              // circular padding
  y[idx] = x[((size_t)b * L + ls) * C + c];
}

__global__ void repack_conv_w(const float* __restrict__ w, float* __restrict__ y,
                              int Cin, int total) {
  // w[o][i][k] (OIW) -> y[o][k*Cin + i]
  int idx = blockIdx.x * blockDim.x + threadIdx.x;
  if (idx >= total) return;
  int k = idx % 3;
  int i = (idx / 3) % Cin;
  int o = idx / (3 * Cin);
  y[(size_t)o * 3 * Cin + k * Cin + i] = w[idx];
}

__global__ void scale_elu(float* __restrict__ x, const float* __restrict__ gamma,
                          const float* __restrict__ beta, int total) {
  int idx = blockIdx.x * blockDim.x + threadIdx.x;
  if (idx >= total) return;
  int c = idx & (DMODEL - 1);
  float v = x[idx] * 0.99999500003749969f * gamma[c] + beta[c]; // 1/sqrt(1+1e-5)
  x[idx] = v > 0.f ? v : (__expf(v) - 1.f);
}

__global__ void maxpool3s2(const float* __restrict__ x, float* __restrict__ y,
                           int L, int total) {
  int idx = blockIdx.x * blockDim.x + threadIdx.x;
  if (idx >= total) return;
  int Lo = L >> 1;
  int c  = idx & (DMODEL - 1);
  int lo = (idx >> 9) % Lo;
  int b  = idx / (Lo * DMODEL);
  float m = -3.0e38f;
#pragma unroll
  for (int d = -1; d <= 1; ++d) {
    int li = 2 * lo + d;
    if (li >= 0 && li < L) m = fmaxf(m, x[((size_t)b * L + li) * DMODEL + c]);
  }
  y[idx] = m;
}

__global__ void slice_out(const float* __restrict__ x, float* __restrict__ out,
                          int total) {
  // x: [B*512, 32] -> out: [B, 256, 32] (last PRED_LEN positions)
  int idx = blockIdx.x * blockDim.x + threadIdx.x;
  if (idx >= total) return;
  int c = idx % 32;
  int t = (idx / 32) % 256;
  int b = idx / (256 * 32);
  out[idx] = x[((size_t)b * 512 + 256 + t) * 32 + c];
}

// ===========================================================================
// Host orchestration
// ===========================================================================
struct AttnP { const float *bk, *bo, *bq, *bv, *wk, *wo, *wq, *wv; };

static inline void launch_gemm(hipStream_t s, const float* A, const float* W,
                               const float* bias, float* C,
                               int M, int N, int K, int act) {
  dim3 g((N + 63) / 64, (M + 127) / 128);
  gemm_wmma<<<g, 256, 0, s>>>(A, W, bias, C, M, N, K, act);
}
static inline void launch_attn(hipStream_t s, const float* Q, const float* K,
                               const float* V, float* O, int Lq, int S, int causal) {
  dim3 g(Lq / 64, NB * NHEADS);
  attn_dense<<<g, 128, 0, s>>>(Q, K, V, O, Lq, S, causal);
}
static inline void launch_ln(hipStream_t s, const float* X, const float* R,
                             const float* g, const float* b, float* Y, int M) {
  layernorm_res<<<(M + 3) / 4, 128, 0, s>>>(X, R, g, b, Y, M);
}
static inline dim3 ew(int total) { return dim3((total + 255) / 256); }

extern "C" void kernel_launch(void* const* d_in, const int* in_sizes, int n_in,
                              void* d_out, int out_size, void* d_ws, size_t ws_size,
                              hipStream_t stream) {
  (void)in_sizes; (void)n_in; (void)out_size; (void)ws_size;

  // -------- input leaf table: jax.tree_leaves order (dict keys sorted) ------
  // top: params, x_dec, x_enc
  // params: conv_layers[2]{b,beta,gamma,w}, dec_emb_w, dec_layers[2]{b1,b2,
  //   cross{bk,bo,bq,bv,wk,wo,wq,wv}, n1b,n1g,n2b,n2g,n3b,n3g,
  //   self{...}, w1,w2}, dec_nb, dec_ng, enc_emb_w,
  //   enc_layers[3]{attn{bk,bo,bq,bv,wk,wo,wq,wv}, b1,b2, n1b,n1g,n2b,n2g,
  //   w1,w2}, enc_nb, enc_ng, proj_b, proj_w
  auto F = [&](int i) { return (const float*)d_in[i]; };
  int p = 0;
  struct ConvP { const float *b, *beta, *gamma, *w; } conv[2];
  for (int l = 0; l < 2; ++l) { conv[l].b = F(p++); conv[l].beta = F(p++);
                                conv[l].gamma = F(p++); conv[l].w = F(p++); }
  const float* dec_emb_w = F(p++);
  struct DecL { const float *b1, *b2; AttnP cross;
                const float *n1b, *n1g, *n2b, *n2g, *n3b, *n3g;
                AttnP self; const float *w1, *w2; } dec[2];
  auto readAttn = [&](AttnP& a) { a.bk = F(p++); a.bo = F(p++); a.bq = F(p++);
    a.bv = F(p++); a.wk = F(p++); a.wo = F(p++); a.wq = F(p++); a.wv = F(p++); };
  for (int l = 0; l < 2; ++l) {
    dec[l].b1 = F(p++); dec[l].b2 = F(p++);
    readAttn(dec[l].cross);
    dec[l].n1b = F(p++); dec[l].n1g = F(p++); dec[l].n2b = F(p++);
    dec[l].n2g = F(p++); dec[l].n3b = F(p++); dec[l].n3g = F(p++);
    readAttn(dec[l].self);
    dec[l].w1 = F(p++); dec[l].w2 = F(p++);
  }
  const float* dec_nb = F(p++); const float* dec_ng = F(p++);
  const float* enc_emb_w = F(p++);
  struct EncL { AttnP attn; const float *b1, *b2;
                const float *n1b, *n1g, *n2b, *n2g; const float *w1, *w2; } enc[3];
  for (int l = 0; l < 3; ++l) {
    readAttn(enc[l].attn);
    enc[l].b1 = F(p++); enc[l].b2 = F(p++);
    enc[l].n1b = F(p++); enc[l].n1g = F(p++); enc[l].n2b = F(p++); enc[l].n2g = F(p++);
    enc[l].w1 = F(p++); enc[l].w2 = F(p++);
  }
  const float* enc_nb = F(p++); const float* enc_ng = F(p++);
  const float* proj_b = F(p++); const float* proj_w = F(p++);
  const float* x_dec = F(p++);
  const float* x_enc = F(p++);

  // -------- workspace layout (~168 MB, fits L2) -----------------------------
  float* ws = (float*)d_ws;
  size_t off = 0;
  auto alloc = [&](size_t n) { float* r = ws + off; off += (n + 63) & ~(size_t)63; return r; };
  float* bufX  = alloc(8192 * 512);   // activations
  float* bufT1 = alloc(8192 * 512);   // attention out / ffn out
  float* bufT2 = alloc(8192 * 512);   // o-proj out
  float* bufQ  = alloc(8192 * 512);
  float* bufK  = alloc(8192 * 512);
  float* bufV  = alloc(8192 * 512);
  float* bufH  = alloc(8192 * 2048);  // ffn hidden / im2col scratch
  float* bufW  = alloc(512 * 1536);   // repacked conv weights
  float* bufEnc = alloc(2048 * 512);  // final encoder output (B*256 rows)

  const hipStream_t s = stream;

  // ======================= ENCODER =========================================
  // token embedding: circular conv1d (C=32,K=3) as im2col + GEMM, + pos embed
  repack_conv_w<<<ew(512 * 32 * 3), 256, 0, s>>>(enc_emb_w, bufW, 32, 512 * 32 * 3);
  im2col_circ<<<ew(8192 * 96), 256, 0, s>>>(x_enc, bufH, 1024, 32, 8192 * 96);
  launch_gemm(s, bufH, bufW, nullptr, bufX, 8192, 512, 96, 0);
  add_pos_embed<<<ew(8192 * 512), 256, 0, s>>>(bufX, 8192 * 512, 1024);

  int L = 1024;
  for (int l = 0; l < 3; ++l) {
    const int M = NB * L;
    const EncL& E = enc[l];
    launch_gemm(s, bufX, E.attn.wq, E.attn.bq, bufQ, M, 512, 512, 0);
    launch_gemm(s, bufX, E.attn.wk, E.attn.bk, bufK, M, 512, 512, 0);
    launch_gemm(s, bufX, E.attn.wv, E.attn.bv, bufV, M, 512, 512, 0);
    launch_attn(s, bufQ, bufK, bufV, bufT1, L, L, 0);
    launch_gemm(s, bufT1, E.attn.wo, E.attn.bo, bufT2, M, 512, 512, 0);
    launch_ln(s, bufX, bufT2, E.n1g, E.n1b, bufX, M);
    launch_gemm(s, bufX, E.w1, E.b1, bufH, M, 2048, 512, 1);   // GELU fused
    launch_gemm(s, bufH, E.w2, E.b2, bufT1, M, 512, 2048, 0);
    launch_ln(s, bufX, bufT1, E.n2g, E.n2b, bufX, M);
    if (l < 2) {  // distilling conv + BN-ish scale + ELU + maxpool3 stride2
      const ConvP& Cv = conv[l];
      repack_conv_w<<<ew(512 * 512 * 3), 256, 0, s>>>(Cv.w, bufW, 512, 512 * 512 * 3);
      im2col_circ<<<ew(M * 1536), 256, 0, s>>>(bufX, bufH, L, 512, M * 1536);
      launch_gemm(s, bufH, bufW, Cv.b, bufT1, M, 512, 1536, 0);
      scale_elu<<<ew(M * 512), 256, 0, s>>>(bufT1, Cv.gamma, Cv.beta, M * 512);
      maxpool3s2<<<ew((M / 2) * 512), 256, 0, s>>>(bufT1, bufX, L, (M / 2) * 512);
      L >>= 1;
    }
  }
  launch_ln(s, bufX, nullptr, enc_ng, enc_nb, bufEnc, NB * L);  // L == 256
  const int S_ENC = L;

  // ======================= DECODER =========================================
  repack_conv_w<<<ew(512 * 32 * 3), 256, 0, s>>>(dec_emb_w, bufW, 32, 512 * 32 * 3);
  im2col_circ<<<ew(4096 * 96), 256, 0, s>>>(x_dec, bufH, 512, 32, 4096 * 96);
  launch_gemm(s, bufH, bufW, nullptr, bufX, 4096, 512, 96, 0);
  add_pos_embed<<<ew(4096 * 512), 256, 0, s>>>(bufX, 4096 * 512, 512);

  const int MD = 4096;          // B * L_DEC
  const int ME = NB * S_ENC;    // encoder rows
  for (int l = 0; l < 2; ++l) {
    const DecL& D = dec[l];
    // masked self-attention
    launch_gemm(s, bufX, D.self.wq, D.self.bq, bufQ, MD, 512, 512, 0);
    launch_gemm(s, bufX, D.self.wk, D.self.bk, bufK, MD, 512, 512, 0);
    launch_gemm(s, bufX, D.self.wv, D.self.bv, bufV, MD, 512, 512, 0);
    launch_attn(s, bufQ, bufK, bufV, bufT1, 512, 512, 1);
    launch_gemm(s, bufT1, D.self.wo, D.self.bo, bufT2, MD, 512, 512, 0);
    launch_ln(s, bufX, bufT2, D.n1g, D.n1b, bufX, MD);
    // cross-attention vs encoder output
    launch_gemm(s, bufX, D.cross.wq, D.cross.bq, bufQ, MD, 512, 512, 0);
    launch_gemm(s, bufEnc, D.cross.wk, D.cross.bk, bufK, ME, 512, 512, 0);
    launch_gemm(s, bufEnc, D.cross.wv, D.cross.bv, bufV, ME, 512, 512, 0);
    launch_attn(s, bufQ, bufK, bufV, bufT1, 512, S_ENC, 0);
    launch_gemm(s, bufT1, D.cross.wo, D.cross.bo, bufT2, MD, 512, 512, 0);
    launch_ln(s, bufX, bufT2, D.n2g, D.n2b, bufX, MD);
    // ffn
    launch_gemm(s, bufX, D.w1, D.b1, bufH, MD, 2048, 512, 1);
    launch_gemm(s, bufH, D.w2, D.b2, bufT1, MD, 512, 2048, 0);
    launch_ln(s, bufX, bufT1, D.n3g, D.n3b, bufX, MD);
  }
  launch_ln(s, bufX, nullptr, dec_ng, dec_nb, bufT1, MD);
  launch_gemm(s, bufT1, proj_w, proj_b, bufT2, MD, 32, 512, 0);
  slice_out<<<ew(8 * 256 * 32), 256, 0, s>>>(bufT2, (float*)d_out, 8 * 256 * 32);
}
